// Seq2Seq_73658689126855
// MI455X (gfx1250) — matrix operand order repaired and checked by reading, not compiled, not run on hardware
//
#include <hip/hip_runtime.h>
#include <hip/hip_bf16.h>
#include <math.h>

// ---------------------------------------------------------------------------
// Seq2Seq LSTM (B=1, S=T=2048, V=50257, D=512, H=3, L=7)
//
//  1) WMMA GEMM: encoder layer-0 input gates for all 2048 steps at once.
//  2) WMMA GEMM: decoder layer-0 input gates for the WHOLE VOCAB (tgt_emb+PE0).
//  3) Pack classifier (w0,w1,w2,b) into a float4 table (1 b128 load / logit).
//  4) Single-wave encoder recurrence (12 lanes = 12 gates, LDS exchange).
//  5) Single-block decoder recurrence: wave0 runs the 7 LSTM cells, 32 waves
//     do the vocab argmax with wave32 shfl_xor butterflies (3 barriers/step).
// ---------------------------------------------------------------------------

typedef __attribute__((ext_vector_type(16))) _Float16 v16h;
typedef __attribute__((ext_vector_type(8)))  float    v8f;

#define DD   512
#define NG   12      // 4*H gates
#define NL   7       // layers
#define HH   3       // hidden
#define SS   2048
#define TT   2048
#define VV   50257
#define VPAD 50272   // 16-aligned

__device__ __forceinline__ float sigmoidf(float x) {
    return 1.0f / (1.0f + __expf(-x));
}

// PE(t, k): even k -> sin(t*div), odd k -> cos(t*div), div = exp(-(k&~1)*ln(1e4)/512)
__device__ __forceinline__ float pe_val(int t, int k) {
    const float c = 0.0179889462f;   // ln(10000)/512
    float e = __expf(-(float)(k & ~1) * c);
    float a = (float)t * e;
    return (k & 1) ? __cosf(a) : __sinf(a);
}

// ---------------------------------------------------------------------------
// GEMM: Gout[m, g] = sum_k (emb[row(m)][k] + PE(t(m), k)) * W[g][k]
//   mode 0 (encoder): row = toks[m], t = m
//   mode 1 (decoder): row = m,       t = 0   (x_t = tgt_emb[tok] + PE[0])
// One wave per 16-row tile; f16 WMMA with hi/lo split (3 wmma per k-block)
// for ~f32 accuracy.  N padded 12 -> 16, Gout stride 16.
// ---------------------------------------------------------------------------
__global__ __launch_bounds__(256) void gates_gemm_kernel(
    const float* __restrict__ emb,     // (V, 512)
    const int*   __restrict__ toks,    // (M,) used in mode 0
    const float* __restrict__ w_ih0,   // (12, 512)
    float*       __restrict__ Gout,    // (Mpad, 16)
    int M, int mode)
{
    int lane = threadIdx.x & 31;
    int wave = threadIdx.x >> 5;
    int tile = blockIdx.x * 8 + wave;
    int Mpad = (M + 15) & ~15;
    int m0 = tile * 16;
    if (m0 >= Mpad) return;

    int myM = m0 + (lane & 15);
    bool mvalid = (myM < M);
    int row = 0, t = 0;
    if (mvalid) {
        if (mode == 0) { row = toks[myM]; t = myM; }
        else           { row = myM;       t = 0;   }
    }
    const float* arow = emb + (size_t)row * DD;
    int n = lane & 15;
    int hiHalf = (lane & 16) ? 1 : 0;

    v8f acc = {};

    for (int kb = 0; kb < DD / 32; ++kb) {
        int k0 = kb * 32;
        v16h a_hi, a_lo, b_hi, b_lo;
#pragma unroll
        for (int i = 0; i < 16; ++i) {
            // ---- A fragment (16x32 f16 layout, ISA 7.12.2) ----
            int r = i >> 1, p = i & 1;
            int kA = ((r & 4) ? 16 : 0) + (hiHalf ? 8 : 0) + ((r & 3) << 1) + p;
            float av = 0.0f;
            if (mvalid) {
                int k = k0 + kA;
                av = arow[k] + pe_val(t, k);
            }
            _Float16 ah = (_Float16)av;
            a_hi[i] = ah;
            a_lo[i] = (_Float16)(av - (float)ah);

            // ---- B fragment (32x16, column-major per lane) ----
            int kB = ((i & 8) ? 16 : 0) + (hiHalf ? 8 : 0) + (i & 7);
            float bv = (n < NG) ? w_ih0[n * DD + k0 + kB] : 0.0f;
            _Float16 bh = (_Float16)bv;
            b_hi[i] = bh;
            b_lo[i] = (_Float16)(bv - (float)bh);
        }
        acc = __builtin_amdgcn_wmma_f32_16x16x32_f16(false, a_hi, false, b_hi,
                                                     (short)0, acc, false, false);
        acc = __builtin_amdgcn_wmma_f32_16x16x32_f16(false, a_hi, false, b_lo,
                                                     (short)0, acc, false, false);
        acc = __builtin_amdgcn_wmma_f32_16x16x32_f16(false, a_lo, false, b_hi,
                                                     (short)0, acc, false, false);
    }

    // ---- store C/D (16x16 f32 layout: lane = N + 16*(M>=8), vgpr r = M&7) ----
#pragma unroll
    for (int r = 0; r < 8; ++r) {
        int Mi = m0 + r + (hiHalf ? 8 : 0);
        if (Mi < M) Gout[(size_t)Mi * 16 + n] = acc[r];
    }
}

// ---------------------------------------------------------------------------
// Pack classifier into float4 rows: clsP[v] = (w0, w1, w2, b)
// ---------------------------------------------------------------------------
__global__ __launch_bounds__(256) void pack_cls_kernel(
    const float* __restrict__ cls_w,   // (V, 3)
    const float* __restrict__ cls_b,   // (V,)
    float4*      __restrict__ clsP,    // (VPAD,)
    int V)
{
    int v = blockIdx.x * 256 + threadIdx.x;
    if (v < V) {
        const float* wv = cls_w + (size_t)v * 3;
        clsP[v] = make_float4(wv[0], wv[1], wv[2], cls_b[v]);
    }
}

// ---------------------------------------------------------------------------
// Encoder recurrence: 2048 steps x 7 layers.  One wave; lane g (<12) owns gate g.
// ---------------------------------------------------------------------------
__global__ __launch_bounds__(32) void enc_seq_kernel(
    const float* __restrict__ G,          // (2048, 16)
    const float* __restrict__ w_ih_rest,  // (6, 12, 3)
    const float* __restrict__ w_hh,       // (7, 12, 3)
    const float* __restrict__ b_ih,       // (7, 12)
    const float* __restrict__ b_hh,       // (7, 12)
    float*       __restrict__ state_out,  // 42: h(7,3) then c(7,3)
    int steps)
{
    __shared__ volatile float pre[NG];
    __shared__ volatile float hS[NL][HH];
    __shared__ volatile float cS[NL][HH];

    int lane = threadIdx.x;
    if (lane < NL * HH) { hS[lane / HH][lane % HH] = 0.0f; cS[lane / HH][lane % HH] = 0.0f; }

    int g = (lane < NG) ? lane : 0;
    float wih[NL][HH], whh[NL][HH], bsum[NL];
#pragma unroll
    for (int l = 0; l < NL; ++l) {
        bsum[l] = b_ih[l * NG + g] + b_hh[l * NG + g];
#pragma unroll
        for (int j = 0; j < HH; ++j) {
            whh[l][j] = w_hh[(l * NG + g) * HH + j];
            wih[l][j] = (l == 0) ? 0.0f : w_ih_rest[((l - 1) * NG + g) * HH + j];
        }
    }

    for (int t = 0; t < steps; ++t) {
#pragma unroll
        for (int l = 0; l < NL; ++l) {
            float p = bsum[l];
            if (l == 0) {
                p += G[(size_t)t * 16 + g];
            } else {
#pragma unroll
                for (int j = 0; j < HH; ++j) p += wih[l][j] * hS[l - 1][j];
            }
#pragma unroll
            for (int j = 0; j < HH; ++j) p += whh[l][j] * hS[l][j];
            if (lane < NG) pre[lane] = p;
            if (lane < HH) {
                float i_ = pre[lane], f_ = pre[lane + 3];
                float g_ = pre[lane + 6], o_ = pre[lane + 9];
                float cv = sigmoidf(f_) * cS[l][lane] + sigmoidf(i_) * tanhf(g_);
                float hv = sigmoidf(o_) * tanhf(cv);
                cS[l][lane] = cv;
                hS[l][lane] = hv;
            }
        }
    }
    if (lane < NL * HH) {
        state_out[lane]           = hS[lane / HH][lane % HH];
        state_out[NL * HH + lane] = cS[lane / HH][lane % HH];
    }
}

// ---------------------------------------------------------------------------
// Decoder recurrence + per-step vocab argmax.  1 block, 1024 threads (32 waves).
// ---------------------------------------------------------------------------
__global__ __launch_bounds__(1024) void dec_seq_kernel(
    const float*  __restrict__ Gv,         // (VPAD, 16) vocab layer-0 gates (incl PE0)
    const float*  __restrict__ w_ih_rest,  // (6, 12, 3)
    const float*  __restrict__ w_hh,       // (7, 12, 3)
    const float*  __restrict__ b_ih,       // (7, 12)
    const float*  __restrict__ b_hh,       // (7, 12)
    const float4* __restrict__ clsP,       // (VPAD,) packed (w0,w1,w2,b)
    const int*    __restrict__ y,          // (T,)
    const int*    __restrict__ tf,         // (T,)
    const float*  __restrict__ state_in,   // 42
    float*        __restrict__ preds,      // (T,)
    int T, int V)
{
    __shared__ volatile float pre[NG];
    __shared__ volatile float hS[NL][HH];
    __shared__ volatile float cS[NL][HH];
    __shared__ float hout[HH];
    __shared__ int   tokS;
    __shared__ float redV[32];
    __shared__ int   redI[32];

    int tid  = threadIdx.x;
    int lane = tid & 31;
    int wid  = tid >> 5;

    float wih[NL][HH], whh[NL][HH], bsum[NL];
    if (tid < 32) {
        int g = (tid < NG) ? tid : 0;
#pragma unroll
        for (int l = 0; l < NL; ++l) {
            bsum[l] = b_ih[l * NG + g] + b_hh[l * NG + g];
#pragma unroll
            for (int j = 0; j < HH; ++j) {
                whh[l][j] = w_hh[(l * NG + g) * HH + j];
                wih[l][j] = (l == 0) ? 0.0f : w_ih_rest[((l - 1) * NG + g) * HH + j];
            }
        }
    }
    if (tid < NL * HH) {
        hS[tid / HH][tid % HH] = state_in[tid];
        cS[tid / HH][tid % HH] = state_in[NL * HH + tid];
    }
    if (tid == 0) tokS = y[T - 1];
    __syncthreads();

    for (int t = 0; t < T; ++t) {
        // ---- wave 0: 7 LSTM cells ----
        if (tid < 32) {
            int g = (tid < NG) ? tid : 0;
            int tok = tokS;
#pragma unroll
            for (int l = 0; l < NL; ++l) {
                float p = bsum[l];
                if (l == 0) {
                    p += Gv[(size_t)tok * 16 + g];
                } else {
#pragma unroll
                    for (int j = 0; j < HH; ++j) p += wih[l][j] * hS[l - 1][j];
                }
#pragma unroll
                for (int j = 0; j < HH; ++j) p += whh[l][j] * hS[l][j];
                if (tid < NG) pre[tid] = p;
                if (tid < HH) {
                    float i_ = pre[tid], f_ = pre[tid + 3];
                    float g_ = pre[tid + 6], o_ = pre[tid + 9];
                    float cv = sigmoidf(f_) * cS[l][tid] + sigmoidf(i_) * tanhf(g_);
                    float hv = sigmoidf(o_) * tanhf(cv);
                    cS[l][tid] = cv;
                    hS[l][tid] = hv;
                }
            }
            if (tid < HH) hout[tid] = hS[NL - 1][tid];
        }
        __syncthreads();   // (1) hout visible to all waves

        // ---- all 1024 threads: argmax over V logits (1 b128 load each) ----
        float h0 = hout[0], h1 = hout[1], h2 = hout[2];
        float bestv = -__builtin_inff();
        int   besti = 0;
        for (int v = tid; v < V; v += 1024) {
            float4 c4 = clsP[v];
            float lv = fmaf(h0, c4.x, fmaf(h1, c4.y, fmaf(h2, c4.z, c4.w)));
            if (lv > bestv) { bestv = lv; besti = v; }   // strict > keeps lowest index
        }
        // wave32 butterfly reduce (lexicographic: val desc, idx asc)
#pragma unroll
        for (int m = 16; m > 0; m >>= 1) {
            float ov = __shfl_xor(bestv, m, 32);
            int   oi = __shfl_xor(besti, m, 32);
            if (ov > bestv || (ov == bestv && oi < besti)) { bestv = ov; besti = oi; }
        }
        if (lane == 0) { redV[wid] = bestv; redI[wid] = besti; }
        __syncthreads();   // (2) per-wave bests visible

        if (tid < 32) {    // wave 0: final 32-way reduce
            float bv = redV[lane];
            int   bi = redI[lane];
#pragma unroll
            for (int m = 16; m > 0; m >>= 1) {
                float ov = __shfl_xor(bv, m, 32);
                int   oi = __shfl_xor(bi, m, 32);
                if (ov > bv || (ov == bv && oi < bi)) { bv = ov; bi = oi; }
            }
            if (lane == 0) {
                preds[t] = (float)bi;
                tokS = (tf[t] > 0) ? y[t] : bi;
            }
        }
        __syncthreads();   // (3) tokS visible for next step
    }
}

// ---------------------------------------------------------------------------
extern "C" void kernel_launch(void* const* d_in, const int* in_sizes, int n_in,
                              void* d_out, int out_size, void* d_ws, size_t ws_size,
                              hipStream_t stream) {
    const int*   x             = (const int*)  d_in[0];
    const int*   y             = (const int*)  d_in[1];
    const int*   tf_mask       = (const int*)  d_in[2];
    const float* src_emb       = (const float*)d_in[3];
    const float* tgt_emb       = (const float*)d_in[4];
    const float* enc_w_ih0     = (const float*)d_in[5];
    const float* enc_w_ih_rest = (const float*)d_in[6];
    const float* enc_w_hh      = (const float*)d_in[7];
    const float* enc_b_ih      = (const float*)d_in[8];
    const float* enc_b_hh      = (const float*)d_in[9];
    const float* dec_w_ih0     = (const float*)d_in[10];
    const float* dec_w_ih_rest = (const float*)d_in[11];
    const float* dec_w_hh      = (const float*)d_in[12];
    const float* dec_b_ih      = (const float*)d_in[13];
    const float* dec_b_hh      = (const float*)d_in[14];
    const float* cls_w         = (const float*)d_in[15];
    const float* cls_b         = (const float*)d_in[16];
    float* out = (float*)d_out;

    // workspace layout (16B aligned slots)
    char*   ws    = (char*)d_ws;
    float*  G_enc = (float*)(ws);                        // 2048*16*4   = 131072 B
    float*  state = (float*)(ws + 131072);               // 42*4        = 168 B
    float*  G_voc = (float*)(ws + 131328);               // 50272*16*4  = 3217408 B
    float4* clsP  = (float4*)(ws + 131328 + 3217408);    // 50272*16    = 804352 B

    // 1) encoder layer-0 gates for all 2048 steps: 128 tiles / 8 waves per block
    gates_gemm_kernel<<<16, 256, 0, stream>>>(src_emb, x, enc_w_ih0, G_enc, SS, 0);
    // 2) decoder layer-0 gates for the whole vocab: 3142 tiles -> 393 blocks
    gates_gemm_kernel<<<(VPAD / 16 + 7) / 8, 256, 0, stream>>>(tgt_emb, x, dec_w_ih0,
                                                               G_voc, VV, 1);
    // 3) pack classifier table
    pack_cls_kernel<<<(VV + 255) / 256, 256, 0, stream>>>(cls_w, cls_b, clsP, VV);
    // 4) encoder recurrence (single wave)
    enc_seq_kernel<<<1, 32, 0, stream>>>(G_enc, enc_w_ih_rest, enc_w_hh,
                                         enc_b_ih, enc_b_hh, state, SS);
    // 5) decoder recurrence + argmax (one workgroup, 32 waves)
    dec_seq_kernel<<<1, 1024, 0, stream>>>(G_voc, dec_w_ih_rest, dec_w_hh,
                                           dec_b_ih, dec_b_hh, clsP,
                                           y, tf_mask, state, out, TT, VV);
}
